// AutoregressiveMixerBlock_61280593379793
// MI455X (gfx1250) — matrix operand order repaired and checked by reading, not compiled, not run on hardware
//
#include <hip/hip_runtime.h>
#include <math.h>

#define B_   16
#define N_   8192
#define H_   128
#define TOK_ 256
#define CH_  512
#define EPS_ 1e-5f
#define NCHUNK 64
#define CHLEN  (N_ / NCHUNK)   // 128

typedef __attribute__((ext_vector_type(16))) __bf16 v16bf;
typedef __attribute__((ext_vector_type(8)))  float  v8f;

// ---------- helpers ----------

static __device__ inline unsigned short f2bf(float f) {
  unsigned u = __float_as_uint(f);
  unsigned r = u + 0x7FFFu + ((u >> 16) & 1u);   // round-to-nearest-even
  return (unsigned short)(r >> 16);
}

static __device__ inline float gelu_exact(float x) {
  return 0.5f * x * (1.0f + erff(x * 0.70710678118654752440f));
}

static __device__ inline v8f vzero8() {
  v8f z;
#pragma unroll
  for (int i = 0; i < 8; ++i) z[i] = 0.0f;
  return z;
}

static __device__ inline v8f wmma_bf16(v16bf a, v16bf b, v8f c) {
  // D = A(16x32) x B(32x16) + C, f32 accumulate
  return __builtin_amdgcn_wmma_f32_16x16x32_bf16(
      false, a, false, b, (short)0, c, false, false);
}

// A fragment from LDS tile row-major [m][k] (ushort, stride mult of 8).
// ISA 16-bit A 16x32: lane m = lane%16; half selects +8 in each K-group of 16;
// VGPR i -> k = (i/4)*16 + half*8 + (i%4)*2 (packed pair). 16B-contiguous per
// 4-reg group -> compiler merges into ds_load_b128.
static __device__ inline v16bf load_a_frag(const unsigned short* base, int stride,
                                           int mbase, int kbase, int lane) {
  union { v16bf v; unsigned u[8]; } f;
  const unsigned short* row = base + (size_t)(mbase + (lane & 15)) * stride;
  int halfk = (lane >> 4) * 8;
#pragma unroll
  for (int i = 0; i < 8; ++i) {
    int k = kbase + ((i >> 2) * 16) + halfk + ((i & 3) * 2);
    f.u[i] = *(const unsigned int*)(row + k);
  }
  return f.v;
}

// B fragment from LDS tile stored TRANSPOSED [n][k].
// 16-bit B 32x16: lane n = lane%16; half selects +8 in each K-group of 16;
// VGPR i -> k = (i/4)*16 + half*8 + (i%4)*2 (packed pair).
static __device__ inline v16bf load_b_frag(const unsigned short* base, int stride,
                                           int nbase, int kbase, int lane) {
  union { v16bf v; unsigned u[8]; } f;
  const unsigned short* row = base + (size_t)(nbase + (lane & 15)) * stride;
  int halfk = (lane >> 4) * 8;
#pragma unroll
  for (int i = 0; i < 8; ++i) {
    int k = kbase + ((i >> 2) * 16) + halfk + ((i & 3) * 2);
    f.u[i] = *(const unsigned int*)(row + k);
  }
  return f.v;
}

// ---------- weight convert + transpose: dst[c][r] = bf16(src[r][c]) ----------
__global__ __launch_bounds__(256) void k_convT(
    const float* __restrict__ src, unsigned short* __restrict__ dst,
    int R, int C) {
  __shared__ float tile[32][33];
  int cb = blockIdx.x * 32, rb = blockIdx.y * 32;
  int tx = threadIdx.x, ty = threadIdx.y;
#pragma unroll
  for (int i = 0; i < 4; ++i) {
    int r = rb + ty + i * 8;
    tile[ty + i * 8][tx] = src[(size_t)r * C + cb + tx];
  }
  __syncthreads();
#pragma unroll
  for (int i = 0; i < 4; ++i) {
    int c = cb + ty + i * 8;
    dst[(size_t)c * R + rb + tx] = f2bf(tile[tx][ty + i * 8]);
  }
}

// ---------- LN1 stats ----------
__global__ __launch_bounds__(256) void k_rowstats(
    const float* __restrict__ x, float2* __restrict__ stats) {
  int row  = blockIdx.x * 8 + (threadIdx.x >> 5);
  int lane = threadIdx.x & 31;
  const float* r = x + (size_t)row * H_;
  float s = 0.f, sq = 0.f;
#pragma unroll
  for (int i = 0; i < 4; ++i) {
    float v = r[lane + 32 * i];
    s += v; sq += v * v;
  }
#pragma unroll
  for (int m = 16; m >= 1; m >>= 1) {
    s  += __shfl_xor(s,  m, 32);
    sq += __shfl_xor(sq, m, 32);
  }
  float mean = s * (1.0f / H_);
  float var  = sq * (1.0f / H_) - mean * mean;
  if (lane == 0) stats[row] = make_float2(mean, rsqrtf(var + EPS_));
}

// ---------- reverse cumsum: chunk sums ----------
__global__ __launch_bounds__(128) void k_chunksum(
    const float* __restrict__ x, const float2* __restrict__ stats,
    float* __restrict__ csum) {
  int b = blockIdx.y, c = blockIdx.x, h = threadIdx.x;
  const float*  xb = x + ((size_t)b * N_ + (size_t)c * CHLEN) * H_;
  const float2* st = stats + b * N_ + c * CHLEN;
  float acc = 0.f;
  for (int j = 0; j < CHLEN; ++j) {
    float2 mr = st[j];
    acc += (xb[(size_t)j * H_ + h] - mr.x) * mr.y;
  }
  csum[((size_t)b * NCHUNK + c) * H_ + h] = acc;
}

// ---------- exclusive suffix across chunks ----------
__global__ __launch_bounds__(128) void k_chunkoffs(float* __restrict__ csum) {
  int b = blockIdx.x, h = threadIdx.x;
  float run = 0.f;
  for (int c = NCHUNK - 1; c >= 0; --c) {
    float* p = &csum[((size_t)b * NCHUNK + c) * H_ + h];
    float t = *p; *p = run; run += t;
  }
}

// ---------- in-chunk reverse scan; emit TRANSPOSED bf16 yct[b][h][tok] ----------
__global__ __launch_bounds__(128) void k_suffix(
    const float* __restrict__ x, const float2* __restrict__ stats,
    const float* __restrict__ csum, const float* __restrict__ g,
    const float* __restrict__ beta, unsigned short* __restrict__ yct) {
  __shared__ __align__(16) unsigned short tile[128][136];  // h x token-in-chunk
  int b = blockIdx.y, c = blockIdx.x, h = threadIdx.x;
  float acc = csum[((size_t)b * NCHUNK + c) * H_ + h];
  float gh = g[h], bh = beta[h];
  int j0 = c * CHLEN;
  const float*  xb = x + (size_t)b * N_ * H_;
  const float2* st = stats + b * N_;
  for (int j = j0 + CHLEN - 1; j >= j0; --j) {
    float2 mr = st[j];
    acc += (xb[(size_t)j * H_ + h] - mr.x) * mr.y;
    tile[h][j - j0] = f2bf(gh * acc + bh * (float)(N_ - j));
  }
  __syncthreads();
  unsigned short* yb = yct + (size_t)b * H_ * N_;
#pragma unroll
  for (int i = 0; i < 16; ++i) {
    int linear = i * 128 + threadIdx.x;
    int jc = linear & 15, hh = linear >> 4;
    *(uint4*)(yb + (size_t)hh * N_ + j0 + jc * 8) = *(const uint4*)&tile[hh][jc * 8];
  }
}

// ---------- token GEMM1: Z1 = gelu(Yc^T @ W1 + b1); M=128 K=8192 N=256 ----------
__global__ __launch_bounds__(128) void k_tok_gemm1(
    const unsigned short* __restrict__ yct, const unsigned short* __restrict__ w1t,
    const float* __restrict__ b1, unsigned short* __restrict__ z1) {
  __shared__ __align__(16) unsigned short As[64][72];
  __shared__ __align__(16) unsigned short Bs[64][72];
  int b = blockIdx.z, mb = blockIdx.y * 64, nb = blockIdx.x * 64;
  int tid = threadIdx.x, lane = tid & 31, w = tid >> 5;
  const unsigned short* Ab = yct + (size_t)b * H_ * N_;

  v8f acc[4];
#pragma unroll
  for (int i = 0; i < 4; ++i) acc[i] = vzero8();

  for (int kt = 0; kt < N_; kt += 64) {
#pragma unroll
    for (int i = 0; i < 4; ++i) {
      int linear = i * 128 + tid;
      int kc = linear & 7, mi = linear >> 3;
      *(uint4*)&As[mi][kc * 8] = *(const uint4*)(Ab + (size_t)(mb + mi) * N_ + kt + kc * 8);
    }
#pragma unroll
    for (int i = 0; i < 4; ++i) {
      int linear = i * 128 + tid;
      int kc = linear & 7, ni = linear >> 3;
      *(uint4*)&Bs[ni][kc * 8] = *(const uint4*)(w1t + (size_t)(nb + ni) * N_ + kt + kc * 8);
    }
    __syncthreads();
    // batch ALL fragment loads first so ds-load latency overlaps the WMMAs
    v16bf a[2], bb[2][4];
#pragma unroll
    for (int s = 0; s < 2; ++s) {
      a[s] = load_a_frag(&As[0][0], 72, w * 16, s * 32, lane);
#pragma unroll
      for (int nt = 0; nt < 4; ++nt)
        bb[s][nt] = load_b_frag(&Bs[0][0], 72, nt * 16, s * 32, lane);
    }
#pragma unroll
    for (int s = 0; s < 2; ++s)
#pragma unroll
      for (int nt = 0; nt < 4; ++nt)
        acc[nt] = wmma_bf16(a[s], bb[s][nt], acc[nt]);
    __syncthreads();
  }

  int nl = lane & 15, halfm = (lane >> 4) * 8;
#pragma unroll
  for (int nt = 0; nt < 4; ++nt) {
    int n = nb + nt * 16 + nl;
    float bias = b1[n];
    union { v8f v; float f[8]; } fr; fr.v = acc[nt];
#pragma unroll
    for (int r = 0; r < 8; ++r) {
      int m = mb + w * 16 + halfm + r;
      z1[((size_t)b * H_ + m) * TOK_ + n] = f2bf(gelu_exact(fr.f[r] + bias));
    }
  }
}

// ---------- token GEMM2: T = Z1 @ W2 + b2 -> transposed fp32 [B,N,H] ----------
__global__ __launch_bounds__(128) void k_tok_gemm2(
    const unsigned short* __restrict__ z1, const unsigned short* __restrict__ w2t,
    const float* __restrict__ b2, float* __restrict__ t) {
  __shared__ __align__(16) unsigned short As[64][72];
  __shared__ __align__(16) unsigned short Bs[64][72];
  int b = blockIdx.z, mb = blockIdx.y * 64, nb = blockIdx.x * 64;
  int tid = threadIdx.x, lane = tid & 31, w = tid >> 5;

  v8f acc[4];
#pragma unroll
  for (int i = 0; i < 4; ++i) acc[i] = vzero8();

  for (int kt = 0; kt < TOK_; kt += 64) {
#pragma unroll
    for (int i = 0; i < 4; ++i) {
      int linear = i * 128 + tid;
      int kc = linear & 7, mi = linear >> 3;
      *(uint4*)&As[mi][kc * 8] =
          *(const uint4*)(z1 + ((size_t)b * H_ + mb + mi) * TOK_ + kt + kc * 8);
    }
#pragma unroll
    for (int i = 0; i < 4; ++i) {
      int linear = i * 128 + tid;
      int kc = linear & 7, ni = linear >> 3;
      *(uint4*)&Bs[ni][kc * 8] =
          *(const uint4*)(w2t + (size_t)(nb + ni) * TOK_ + kt + kc * 8);
    }
    __syncthreads();
    v16bf a[2], bb[2][4];
#pragma unroll
    for (int s = 0; s < 2; ++s) {
      a[s] = load_a_frag(&As[0][0], 72, w * 16, s * 32, lane);
#pragma unroll
      for (int nt = 0; nt < 4; ++nt)
        bb[s][nt] = load_b_frag(&Bs[0][0], 72, nt * 16, s * 32, lane);
    }
#pragma unroll
    for (int s = 0; s < 2; ++s)
#pragma unroll
      for (int nt = 0; nt < 4; ++nt)
        acc[nt] = wmma_bf16(a[s], bb[s][nt], acc[nt]);
    __syncthreads();
  }

  int nl = lane & 15, halfm = (lane >> 4) * 8;
#pragma unroll
  for (int nt = 0; nt < 4; ++nt) {
    int n = nb + nt * 16 + nl;          // token index
    float bias = b2[n];
    union { v8f v; float f[8]; } fr; fr.v = acc[nt];
    union { float4 q; float f[4]; } lo, hi;
#pragma unroll
    for (int r = 0; r < 4; ++r) { lo.f[r] = fr.f[r] + bias; hi.f[r] = fr.f[4 + r] + bias; }
    float* base = t + ((size_t)b * N_ + n) * H_ + mb + w * 16 + halfm;
    *(float4*)base       = lo.q;        // 8 contiguous h per lane
    *(float4*)(base + 4) = hi.q;
  }
}

// ---------- LN2 -> bf16 U ----------
__global__ __launch_bounds__(256) void k_ln2(
    const float* __restrict__ t, const float* __restrict__ g,
    const float* __restrict__ beta, unsigned short* __restrict__ u) {
  int row  = blockIdx.x * 8 + (threadIdx.x >> 5);
  int lane = threadIdx.x & 31;
  const float* r = t + (size_t)row * H_;
  float s = 0.f, sq = 0.f, vals[4];
#pragma unroll
  for (int i = 0; i < 4; ++i) {
    float v = r[lane + 32 * i];
    vals[i] = v; s += v; sq += v * v;
  }
#pragma unroll
  for (int m = 16; m >= 1; m >>= 1) {
    s  += __shfl_xor(s,  m, 32);
    sq += __shfl_xor(sq, m, 32);
  }
  float mean = s * (1.0f / H_);
  float rstd = rsqrtf(sq * (1.0f / H_) - mean * mean + EPS_);
#pragma unroll
  for (int i = 0; i < 4; ++i) {
    int idx = lane + 32 * i;
    u[(size_t)row * H_ + idx] = f2bf((vals[i] - mean) * rstd * g[idx] + beta[idx]);
  }
}

// ---------- fused channel MLP ----------
union __align__(16) ChSm {
  struct {
    unsigned short Us[64][136];    // rows x K=128
    unsigned short Bs1[32][136];   // hidden-n x K=128 (transposed)
    unsigned short Bs2[128][40];   // out-n x K=32 (transposed)
    unsigned short Hs[64][40];     // hidden tile, A-layout
  } s;
  float ot[64][132];               // fp32 output bounce (used after loop)
};

__global__ __launch_bounds__(128) void k_channel(
    const unsigned short* __restrict__ u, const unsigned short* __restrict__ w1t,
    const float* __restrict__ b1, const unsigned short* __restrict__ w2t,
    const float* __restrict__ b2, float* __restrict__ out) {
  __shared__ ChSm sm;
  int rowbase = blockIdx.x * 64;
  int tid = threadIdx.x, lane = tid & 31, w = tid >> 5;
  int nl = lane & 15, halfm = (lane >> 4) * 8;

#pragma unroll
  for (int i = 0; i < 8; ++i) {            // stage U tile once
    int linear = i * 128 + tid;
    int kc = linear & 15, mi = linear >> 4;
    *(uint4*)&sm.s.Us[mi][kc * 8] =
        *(const uint4*)(u + (size_t)(rowbase + mi) * H_ + kc * 8);
  }

  v8f oacc[8];
#pragma unroll
  for (int i = 0; i < 8; ++i) oacc[i] = vzero8();

  for (int ch = 0; ch < CH_ / 32; ++ch) {
    __syncthreads();
#pragma unroll
    for (int i = 0; i < 4; ++i) {          // W1^T chunk: 32 x 128
      int linear = i * 128 + tid;
      int kc = linear & 15, ni = linear >> 4;
      *(uint4*)&sm.s.Bs1[ni][kc * 8] =
          *(const uint4*)(w1t + (size_t)(ch * 32 + ni) * H_ + kc * 8);
    }
#pragma unroll
    for (int i = 0; i < 4; ++i) {          // W2^T chunk: 128 x 32
      int linear = i * 128 + tid;
      int kc = linear & 3, ni = linear >> 2;
      *(uint4*)&sm.s.Bs2[ni][kc * 8] =
          *(const uint4*)(w2t + (size_t)ni * CH_ + ch * 32 + kc * 8);
    }
    __syncthreads();

    // hidden = U(64x128) @ W1[:,chunk](128x32); batch fragment loads
    v16bf a1[4], bb1[4][2];
#pragma unroll
    for (int ks = 0; ks < 4; ++ks) {
      a1[ks] = load_a_frag(&sm.s.Us[0][0], 136, w * 16, ks * 32, lane);
#pragma unroll
      for (int nt = 0; nt < 2; ++nt)
        bb1[ks][nt] = load_b_frag(&sm.s.Bs1[0][0], 136, nt * 16, ks * 32, lane);
    }
    v8f hacc[2] = {vzero8(), vzero8()};
#pragma unroll
    for (int ks = 0; ks < 4; ++ks)
#pragma unroll
      for (int nt = 0; nt < 2; ++nt)
        hacc[nt] = wmma_bf16(a1[ks], bb1[ks][nt], hacc[nt]);

    // gelu + stash hidden tile in LDS (A-layout for second GEMM)
#pragma unroll
    for (int nt = 0; nt < 2; ++nt) {
      union { v8f v; float f[8]; } fr; fr.v = hacc[nt];
      float bias = b1[ch * 32 + nt * 16 + nl];
#pragma unroll
      for (int r = 0; r < 8; ++r) {
        int m = w * 16 + halfm + r;
        sm.s.Hs[m][nt * 16 + nl] = f2bf(gelu_exact(fr.f[r] + bias));
      }
    }
    __syncthreads();

    // out += H(64x32) @ W2[chunk,:](32x128); batch fragment loads
    v16bf a2 = load_a_frag(&sm.s.Hs[0][0], 40, w * 16, 0, lane);
    v16bf bb2[8];
#pragma unroll
    for (int nt2 = 0; nt2 < 8; ++nt2)
      bb2[nt2] = load_b_frag(&sm.s.Bs2[0][0], 40, nt2 * 16, 0, lane);
#pragma unroll
    for (int nt2 = 0; nt2 < 8; ++nt2)
      oacc[nt2] = wmma_bf16(a2, bb2[nt2], oacc[nt2]);
  }

  // epilogue: bounce through LDS fp32 tile for coalesced float4 stores
  __syncthreads();
#pragma unroll
  for (int nt2 = 0; nt2 < 8; ++nt2) {
    union { v8f v; float f[8]; } fr; fr.v = oacc[nt2];
    int n = nt2 * 16 + nl;
    float bias = b2[n];
#pragma unroll
    for (int r = 0; r < 8; ++r) sm.ot[w * 16 + halfm + r][n] = fr.f[r] + bias;
  }
  __syncthreads();
#pragma unroll
  for (int i = 0; i < 16; ++i) {
    int linear = i * 128 + tid;
    int nc = linear & 31, mi = linear >> 5;
    *(float4*)(out + (size_t)(rowbase + mi) * H_ + nc * 4) =
        *(const float4*)&sm.ot[mi][nc * 4];
  }
}

// ---------- launch ----------

extern "C" void kernel_launch(void* const* d_in, const int* in_sizes, int n_in,
                              void* d_out, int out_size, void* d_ws, size_t ws_size,
                              hipStream_t stream) {
  (void)in_sizes; (void)n_in; (void)out_size; (void)ws_size;
  const float* x      = (const float*)d_in[0];
  const float* ln1_g  = (const float*)d_in[1];
  const float* ln1_b  = (const float*)d_in[2];
  const float* ln2_g  = (const float*)d_in[3];
  const float* ln2_b  = (const float*)d_in[4];
  const float* tok_w1 = (const float*)d_in[5];
  const float* tok_b1 = (const float*)d_in[6];
  const float* tok_w2 = (const float*)d_in[7];
  const float* tok_b2 = (const float*)d_in[8];
  const float* ch_w1  = (const float*)d_in[9];
  const float* ch_b1  = (const float*)d_in[10];
  const float* ch_w2  = (const float*)d_in[11];
  const float* ch_b2  = (const float*)d_in[12];
  float* out = (float*)d_out;

  char* ws = (char*)d_ws;
  size_t off = 0;
  auto take = [&](size_t bytes) -> char* {
    char* p = ws + off;
    off += (bytes + 255) & ~(size_t)255;
    return p;
  };
  float2*         stats = (float2*)        take((size_t)B_ * N_ * sizeof(float2));
  float*          csum  = (float*)         take((size_t)B_ * NCHUNK * H_ * 4);
  unsigned short* yct   = (unsigned short*)take((size_t)B_ * H_ * N_ * 2);
  unsigned short* w1t   = (unsigned short*)take((size_t)TOK_ * N_ * 2);  // [256][8192]
  unsigned short* w2t   = (unsigned short*)take((size_t)N_ * TOK_ * 2);  // [8192][256]
  unsigned short* z1    = (unsigned short*)take((size_t)B_ * H_ * TOK_ * 2);
  float*          t     = (float*)         take((size_t)B_ * N_ * H_ * 4);
  unsigned short* ub    = (unsigned short*)take((size_t)B_ * N_ * H_ * 2);
  unsigned short* cw1t  = (unsigned short*)take((size_t)CH_ * H_ * 2);   // [512][128]
  unsigned short* cw2t  = (unsigned short*)take((size_t)H_ * CH_ * 2);   // [128][512]

  dim3 tb(32, 8);
  // weights: convert fp32->bf16 AND transpose so every GEMM stage is K-contiguous
  k_convT<<<dim3(TOK_ / 32, N_ / 32), tb, 0, stream>>>(tok_w1, w1t, N_, TOK_);
  k_convT<<<dim3(N_ / 32, TOK_ / 32), tb, 0, stream>>>(tok_w2, w2t, TOK_, N_);
  k_convT<<<dim3(CH_ / 32, H_ / 32),  tb, 0, stream>>>(ch_w1, cw1t, H_, CH_);
  k_convT<<<dim3(H_ / 32, CH_ / 32),  tb, 0, stream>>>(ch_w2, cw2t, CH_, H_);

  // LN1 + reverse cumsum (chunked suffix scan) -> yct (bf16, transposed)
  k_rowstats <<<(B_ * N_) / 8, 256, 0, stream>>>(x, stats);
  k_chunksum <<<dim3(NCHUNK, B_), 128, 0, stream>>>(x, stats, csum);
  k_chunkoffs<<<B_, 128, 0, stream>>>(csum);
  k_suffix   <<<dim3(NCHUNK, B_), 128, 0, stream>>>(x, stats, csum, ln1_g, ln1_b, yct);

  // token MLP (WMMA bf16)
  k_tok_gemm1<<<dim3(TOK_ / 64, H_ / 64, B_), 128, 0, stream>>>(yct, w1t, tok_b1, z1);
  k_tok_gemm2<<<dim3(N_ / 64, H_ / 64, B_), 128, 0, stream>>>(z1, w2t, tok_b2, t);

  // LN2 -> bf16
  k_ln2<<<(B_ * N_) / 8, 256, 0, stream>>>(t, ln2_g, ln2_b, ub);

  // fused channel MLP (WMMA bf16, hidden never leaves LDS)
  k_channel<<<(B_ * N_) / 64, 128, 0, stream>>>(ub, cw1t, ch_b1, cw2t, ch_b2, out);
}
